// Attention_11665131176467
// MI455X (gfx1250) — matrix operand order repaired and checked by reading, not compiled, not run on hardware
//
#include <hip/hip_runtime.h>
#include <stdint.h>

// ---------------- CDNA5 WMMA types ----------------
typedef __attribute__((ext_vector_type(16))) __bf16 v16bf;
typedef __attribute__((ext_vector_type(8)))  float  v8f;
typedef __attribute__((ext_vector_type(4)))  unsigned int u32x4;
typedef __attribute__((ext_vector_type(4)))  int v4i;

union FragAB { v16bf bf; u32x4 q[2]; };

#define DIMX   1024
#define HEADS  16
#define DHEAD  64
#define INNER  1024
#define NBATCH 4
#define NSEQ   2048
#define MTOT   (NBATCH * NSEQ)   // 8192 rows of x flattened
#define SCALEF 0.125f            // 64^-0.5

// -------- CDNA5 async / TDM paths (guarded; fallback = manual staging) ------
#if __has_builtin(__builtin_amdgcn_global_load_async_to_lds_b128) && \
    __has_builtin(__builtin_amdgcn_s_wait_asynccnt)
#define USE_ASYNC_LDS 1
#else
#define USE_ASYNC_LDS 0
#endif

#if __has_builtin(__builtin_amdgcn_tensor_load_to_lds) && \
    __has_builtin(__builtin_amdgcn_s_wait_tensorcnt)
#define USE_TDM 1
#else
#define USE_TDM 0
#endif

// generic -> AS1 (global) / AS3 (LDS) v4i pointer casts.
// (generic LDS address low 32 bits == LDS byte offset per aperture rules)
#define ASG(p) ((__attribute__((address_space(1))) v4i*)(uintptr_t)(p))
#define ASL(p) ((__attribute__((address_space(3))) v4i*)(unsigned)(uintptr_t)(p))

#if USE_TDM
typedef unsigned int tdm_u32x4 __attribute__((ext_vector_type(4)));
typedef int          tdm_i32x8 __attribute__((ext_vector_type(8)));
typedef int          tdm_i32x4 __attribute__((ext_vector_type(4)));

// DMA a 64x64 bf16 tile (row stride 64 elems) from global into LDS.
// D# packing per CDNA5 ISA:
//  g0: [1:0]=count=1, [63:32]=lds_addr, [120:64]=global_addr, [127:126]=type=2
//  g1: data_size=1 (2B) @bits17:16, tensor_dim0=64 @bits79:48,
//      tensor_dim1=2048 @bits111:80, tile_dim0=64 @bits127:112,
//      tile_dim1=64 @bits143:128, tensor_dim0_stride=64 @bits207:160
__device__ __forceinline__ void tdm_load_tile64x64_bf16(const __bf16* gsrc,
                                                        __bf16* lds) {
  unsigned long long ga = (unsigned long long)(uintptr_t)gsrc;
  tdm_u32x4 g0;
  g0.x = 1u;                                  // count=1, user mode
  g0.y = (unsigned)(uintptr_t)lds;            // lds_addr (byte offset)
  g0.z = (unsigned)ga;                        // global_addr[31:0]
  g0.w = (unsigned)(ga >> 32) | (2u << 30);   // global_addr[56:32] | type=2
  tdm_i32x8 g1;
  g1[0] = 0x00010000;                         // data_size = 2 bytes
  g1[1] = (int)(64u << 16);                   // tensor_dim0 = 64 (lo bits)
  g1[2] = (int)(2048u << 16);                 // tensor_dim1 = 2048 (lo bits)
  g1[3] = (int)(64u << 16);                   // tile_dim0 = 64
  g1[4] = 64;                                 // tile_dim1 = 64
  g1[5] = 64;                                 // tensor_dim0_stride = 64
  g1[6] = 0;
  g1[7] = 0;
  tdm_i32x4 z4 = {0, 0, 0, 0};
#if __clang_major__ >= 23
  tdm_i32x8 z8 = {0, 0, 0, 0, 0, 0, 0, 0};
  __builtin_amdgcn_tensor_load_to_lds(g0, g1, z4, z4, z8, 0);
#else
  __builtin_amdgcn_tensor_load_to_lds(g0, g1, z4, z4, 0);
#endif
}
#endif  // USE_TDM

// ---------------- fp32 -> bf16 conversion ----------------
__global__ void cvt_f32_bf16(const float* __restrict__ src,
                             __bf16* __restrict__ dst, int n4) {
  int i = blockIdx.x * blockDim.x + threadIdx.x;
  for (; i < n4; i += gridDim.x * blockDim.x) {
    float4 f = ((const float4*)src)[i];
    __bf16* d = dst + (size_t)i * 4;
    d[0] = (__bf16)f.x; d[1] = (__bf16)f.y;
    d[2] = (__bf16)f.z; d[3] = (__bf16)f.w;
  }
}

// ---------------- bf16 GEMM: C[MTOT,NCOLS] = A[MTOT,1024] @ B[1024,NCOLS] ----
// MODE 0: scatter into q/k/v head-major bf16 buffers [b,h,n,d]
// MODE 1: fp32 output + bias (final projection)
template <int MODE, int NCOLS>
__global__ __launch_bounds__(256) void gemm_bf16(
    const __bf16* __restrict__ A, const __bf16* __restrict__ Bw,
    __bf16* __restrict__ qb, __bf16* __restrict__ kb, __bf16* __restrict__ vb,
    const float* __restrict__ bias, float* __restrict__ Cout) {
  __shared__ __align__(16) __bf16 At[128 * 32];   // [row][k]
  __shared__ __align__(16) __bf16 Bt[128 * 32];   // transposed: [n][k]

  const int tid  = threadIdx.x;
  const int lane = tid & 31;
  const int w    = tid >> 5;          // 8 waves
  const int wm   = w & 3, wn = w >> 2;
  const int half = lane >> 4, ln = lane & 15;
  const int m0 = blockIdx.y * 128;
  const int n0 = blockIdx.x * 128;

  v8f acc[2][4];
  v8f vz = {};
  #pragma unroll
  for (int a = 0; a < 2; a++)
    #pragma unroll
    for (int b = 0; b < 4; b++) acc[a][b] = vz;

  for (int k0 = 0; k0 < DIMX; k0 += 32) {
    // Stage A tile [128][32]
#if USE_ASYNC_LDS
    for (int idx = tid; idx < 512; idx += 256) {
      int r = idx >> 2, c = idx & 3;
      __builtin_amdgcn_global_load_async_to_lds_b128(
          ASG(&A[(size_t)(m0 + r) * DIMX + k0 + c * 8]),
          ASL(&At[r * 32 + c * 8]), 0, 0);
    }
#else
    for (int idx = tid; idx < 512; idx += 256) {
      int r = idx >> 2, c = idx & 3;
      *(u32x4*)&At[r * 32 + c * 8] =
          *(const u32x4*)&A[(size_t)(m0 + r) * DIMX + k0 + c * 8];
    }
#endif
    // Stage B tile transposed -> Bt[n][k]
    for (int idx = tid; idx < 512; idx += 256) {
      int k = idx >> 4, c = idx & 15;
      u32x4 d = *(const u32x4*)&Bw[(size_t)(k0 + k) * NCOLS + n0 + c * 8];
      const __bf16* dv = (const __bf16*)&d;
      #pragma unroll
      for (int j = 0; j < 8; j++) Bt[(c * 8 + j) * 32 + k] = dv[j];
    }
#if USE_ASYNC_LDS
    __builtin_amdgcn_s_wait_asynccnt(0);
#endif
    __syncthreads();

    FragAB af[2];
    #pragma unroll
    for (int mt = 0; mt < 2; mt++) {
      int row = wm * 32 + mt * 16 + ln;
      af[mt].q[0] = *(const u32x4*)&At[row * 32 + half * 8];        // K = half*8 + 0..7
      af[mt].q[1] = *(const u32x4*)&At[row * 32 + 16 + half * 8];   // K = 16+half*8 + 0..7
    }
    #pragma unroll
    for (int nt = 0; nt < 4; nt++) {
      FragAB bfm;
      int col = wn * 64 + nt * 16 + ln;
      bfm.q[0] = *(const u32x4*)&Bt[col * 32 + half * 8];           // K = half*8 + 0..7
      bfm.q[1] = *(const u32x4*)&Bt[col * 32 + 16 + half * 8];      // K = 16+half*8 + 0..7
      #pragma unroll
      for (int mt = 0; mt < 2; mt++)
        acc[mt][nt] = __builtin_amdgcn_wmma_f32_16x16x32_bf16(
            false, af[mt].bf, false, bfm.bf, (short)0, acc[mt][nt], false, false);
    }
    __syncthreads();
  }

  #pragma unroll
  for (int mt = 0; mt < 2; mt++)
    #pragma unroll
    for (int nt = 0; nt < 4; nt++)
      #pragma unroll
      for (int r = 0; r < 8; r++) {
        int m = m0 + wm * 32 + mt * 16 + half * 8 + r;
        int n = n0 + wn * 64 + nt * 16 + ln;
        float v = acc[mt][nt][r];
        if (MODE == 0) {
          int bi = m >> 11, i = m & 2047;
          int which = n >> 10, rem = n & 1023;
          int h = rem >> 6, d = rem & 63;
          __bf16* dst = (which == 0) ? qb : (which == 1) ? kb : vb;
          dst[((size_t)(bi * HEADS + h) * NSEQ + i) * DHEAD + d] = (__bf16)v;
        } else {
          Cout[(size_t)m * NCOLS + n] = v + bias[n];
        }
      }
}

// ---------------- flash attention: per (b,h) 64-query block -----------------
__global__ __launch_bounds__(128) void attn_kernel(
    const __bf16* __restrict__ qb, const __bf16* __restrict__ kb,
    const __bf16* __restrict__ vb, __bf16* __restrict__ ob) {
  __shared__ __align__(16) __bf16 Kt[64 * 64];      // [key][d]
  __shared__ __align__(16) __bf16 Vt[64 * 64];      // transposed [d][key]
  __shared__ __align__(16) __bf16 Pt[4 * 16 * 64];  // per-wave P

  const int tid  = threadIdx.x;
  const int lane = tid & 31;
  const int w    = tid >> 5;        // 4 waves, 16 query rows each
  const int half = lane >> 4, ln = lane & 15;
  const int bh   = blockIdx.z * HEADS + blockIdx.y;
  const int qrow = blockIdx.x * 64 + w * 16;

  const __bf16* qbase = qb + (size_t)bh * NSEQ * DHEAD;
  const __bf16* kbase = kb + (size_t)bh * NSEQ * DHEAD;
  const __bf16* vbase = vb + (size_t)bh * NSEQ * DHEAD;

  // Q fragments for both 32-wide K-steps over d (resident for the whole loop)
  FragAB qf[2];
  #pragma unroll
  for (int kk = 0; kk < 2; kk++) {
    const __bf16* qr = qbase + (size_t)(qrow + ln) * DHEAD + kk * 32 + half * 8;
    qf[kk].q[0] = *(const u32x4*)qr;
    qf[kk].q[1] = *(const u32x4*)(qr + 16);
  }

  v8f vz = {};
  v8f oacc[4];
  float mrow[8], lrow[8];
  #pragma unroll
  for (int t = 0; t < 4; t++) oacc[t] = vz;
  #pragma unroll
  for (int r = 0; r < 8; r++) { mrow[r] = -__builtin_inff(); lrow[r] = 0.f; }

  for (int jt = 0; jt < NSEQ / 64; jt++) {
    // Stage K tile: TDM DMA (wave 0) if available, else cooperative copy
#if USE_TDM
    if (w == 0)
      tdm_load_tile64x64_bf16(kbase + (size_t)(jt * 64) * DHEAD, &Kt[0]);
#else
    for (int idx = tid; idx < 512; idx += 128) {
      int r = idx >> 3, c = idx & 7;
      *(u32x4*)&Kt[r * 64 + c * 8] =
          *(const u32x4*)&kbase[(size_t)(jt * 64 + r) * DHEAD + c * 8];
    }
#endif
    // Stage V tile transposed -> Vt[d][key] (all threads)
    for (int idx = tid; idx < 512; idx += 128) {
      int r = idx >> 3, c = idx & 7;
      u32x4 dv = *(const u32x4*)&vbase[(size_t)(jt * 64 + r) * DHEAD + c * 8];
      const __bf16* pv = (const __bf16*)&dv;
      #pragma unroll
      for (int j = 0; j < 8; j++) Vt[(c * 8 + j) * 64 + r] = pv[j];
    }
#if USE_TDM
    if (w == 0) __builtin_amdgcn_s_wait_tensorcnt(0);
#endif
    __syncthreads();

    // S = Q @ K^T  (B fragment = contiguous d-run from a key row)
    v8f sacc[4];
    #pragma unroll
    for (int t = 0; t < 4; t++) sacc[t] = vz;
    #pragma unroll
    for (int kk = 0; kk < 2; kk++)
      #pragma unroll
      for (int nt = 0; nt < 4; nt++) {
        FragAB bfm;
        int key = nt * 16 + ln;
        bfm.q[0] = *(const u32x4*)&Kt[key * 64 + kk * 32 + half * 8];
        bfm.q[1] = *(const u32x4*)&Kt[key * 64 + kk * 32 + half * 8 + 16];
        sacc[nt] = __builtin_amdgcn_wmma_f32_16x16x32_bf16(
            false, qf[kk].bf, false, bfm.bf, (short)0, sacc[nt], false, false);
      }

    // Online softmax: lane owns rows M = half*8 + r; reduce over N in half-wave
    #pragma unroll
    for (int r = 0; r < 8; r++) {
      float rmax = -__builtin_inff();
      #pragma unroll
      for (int nt = 0; nt < 4; nt++) rmax = fmaxf(rmax, sacc[nt][r]);
      #pragma unroll
      for (int off = 8; off >= 1; off >>= 1)
        rmax = fmaxf(rmax, __shfl_xor(rmax, off, 16));
      rmax *= SCALEF;
      float mnew  = fmaxf(mrow[r], rmax);
      float alpha = __expf(mrow[r] - mnew);
      mrow[r] = mnew;
      float rsum = 0.f;
      #pragma unroll
      for (int nt = 0; nt < 4; nt++) {
        float p = __expf(sacc[nt][r] * SCALEF - mnew);
        sacc[nt][r] = p;
        rsum += p;
      }
      #pragma unroll
      for (int off = 8; off >= 1; off >>= 1)
        rsum += __shfl_xor(rsum, off, 16);
      lrow[r] = lrow[r] * alpha + rsum;
      #pragma unroll
      for (int nt = 0; nt < 4; nt++) oacc[nt][r] *= alpha;
    }

    // P -> per-wave LDS, then re-read as A fragments for O += P @ V
    __bf16* Pw = &Pt[w * 16 * 64];
    #pragma unroll
    for (int nt = 0; nt < 4; nt++)
      #pragma unroll
      for (int r = 0; r < 8; r++)
        Pw[(half * 8 + r) * 64 + nt * 16 + ln] = (__bf16)sacc[nt][r];
    asm volatile("s_wait_dscnt 0" ::: "memory");   // wave-local LDS RAW fence

    #pragma unroll
    for (int kk = 0; kk < 2; kk++) {
      FragAB pf;
      const __bf16* pr = &Pw[ln * 64 + kk * 32 + half * 8];
      pf.q[0] = *(const u32x4*)pr;
      pf.q[1] = *(const u32x4*)(pr + 16);
      #pragma unroll
      for (int nt = 0; nt < 4; nt++) {
        FragAB bfm;
        int dcol = nt * 16 + ln;
        bfm.q[0] = *(const u32x4*)&Vt[dcol * 64 + kk * 32 + half * 8];
        bfm.q[1] = *(const u32x4*)&Vt[dcol * 64 + kk * 32 + half * 8 + 16];
        oacc[nt] = __builtin_amdgcn_wmma_f32_16x16x32_bf16(
            false, pf.bf, false, bfm.bf, (short)0, oacc[nt], false, false);
      }
    }
    __syncthreads();
  }

  // Normalize and write attention output [B*N, INNER] bf16
  const int brow = blockIdx.z * NSEQ + qrow;
  #pragma unroll
  for (int r = 0; r < 8; r++) {
    float inv = 1.0f / lrow[r];
    int grow = brow + half * 8 + r;
    #pragma unroll
    for (int nt = 0; nt < 4; nt++)
      ob[(size_t)grow * INNER + blockIdx.y * DHEAD + nt * 16 + ln] =
          (__bf16)(oacc[nt][r] * inv);
  }
}

// ---------------- host launcher ----------------
extern "C" void kernel_launch(void* const* d_in, const int* in_sizes, int n_in,
                              void* d_out, int out_size, void* d_ws, size_t ws_size,
                              hipStream_t stream) {
  const float* x     = (const float*)d_in[0];
  const float* w_qkv = (const float*)d_in[1];
  const float* w_out = (const float*)d_in[2];
  const float* b_out = (const float*)d_in[3];
  float* out = (float*)d_out;

  char* ws = (char*)d_ws;
  size_t off = 0;
  auto alloc = [&](size_t bytes) {
    void* p = ws + off;
    off += (bytes + 255) & ~(size_t)255;
    return p;
  };
  __bf16* xb   = (__bf16*)alloc((size_t)MTOT * DIMX * 2);
  __bf16* wqb  = (__bf16*)alloc((size_t)DIMX * 3 * INNER * 2);
  __bf16* wob  = (__bf16*)alloc((size_t)INNER * DIMX * 2);
  __bf16* qbuf = (__bf16*)alloc((size_t)MTOT * INNER * 2);
  __bf16* kbuf = (__bf16*)alloc((size_t)MTOT * INNER * 2);
  __bf16* vbuf = (__bf16*)alloc((size_t)MTOT * INNER * 2);
  __bf16* aob  = (__bf16*)alloc((size_t)MTOT * INNER * 2);

  cvt_f32_bf16<<<1024, 256, 0, stream>>>(x, xb, (MTOT * DIMX) / 4);
  cvt_f32_bf16<<<1024, 256, 0, stream>>>(w_qkv, wqb, (DIMX * 3 * INNER) / 4);
  cvt_f32_bf16<<<1024, 256, 0, stream>>>(w_out, wob, (INNER * DIMX) / 4);

  gemm_bf16<0, 3 * INNER><<<dim3((3 * INNER) / 128, MTOT / 128), 256, 0, stream>>>(
      xb, wqb, qbuf, kbuf, vbuf, nullptr, nullptr);

  attn_kernel<<<dim3(NSEQ / 64, HEADS, NBATCH), 128, 0, stream>>>(
      qbuf, kbuf, vbuf, aob);

  gemm_bf16<1, DIMX><<<dim3(DIMX / 128, MTOT / 128), 256, 0, stream>>>(
      aob, wob, nullptr, nullptr, nullptr, b_out, out);
}
